// DyHeadBlock_29901562314840
// MI455X (gfx1250) — compile-verified, matrix operand order checked by reading, not run on hardware
//
#include <hip/hip_runtime.h>
#include <hip/hip_bf16.h>
#include <stdint.h>

// ---------------------------------------------------------------------------
// DyHead block for MI455X (gfx1250, wave32, WMMA).
// Heavy mdcn GEMMs run as fused bilinear-gather + v_wmma_f32_16x16x32_bf16.
// K-step = 64, 2 M-tiles per wave -> 4 WMMAs per sync round.
// ---------------------------------------------------------------------------

typedef __attribute__((ext_vector_type(16))) __bf16 v16bf;
typedef __attribute__((ext_vector_type(8)))  float  v8f;

#define CH 256
#define KTOT 2304   // 9 taps * 256 in-channels, K index = tap*256 + c

__device__ __forceinline__ unsigned short f2bf(float f) {
    unsigned int u = __float_as_uint(f);
    unsigned int r = u + 0x7FFFu + ((u >> 16) & 1u);   // round-to-nearest-even
    return (unsigned short)(r >> 16);
}

__device__ __forceinline__ float hsig(float x) {
    return fminf(fmaxf((x + 3.0f) * (1.0f / 6.0f), 0.0f), 1.0f);
}

// --------------------------------------------------------------------------
// Repack conv weight (O=256, C=256, 3, 3) fp32 -> bf16 [O][K], K = tap*256+c
// --------------------------------------------------------------------------
__global__ __launch_bounds__(256)
void prep_w_kernel(const float* __restrict__ w, unsigned short* __restrict__ wq) {
    int idx = blockIdx.x * 256 + threadIdx.x;
    if (idx >= CH * KTOT) return;
    int o = idx / KTOT;
    int rem = idx - o * KTOT;
    int k = rem >> 8;
    int c = rem & 255;
    wq[idx] = f2bf(w[(o * CH + c) * 9 + k]);
}

// --------------------------------------------------------------------------
// Offset conv: om = conv3x3(x, w_off(27,256,3,3)) + b_off
//   channels 0..17 -> offs, 18..26 -> sigmoid -> mask
// One block (256 threads = one thread per in-channel) per output pixel.
// --------------------------------------------------------------------------
__global__ __launch_bounds__(256)
void offset_conv_kernel(const float* __restrict__ x,
                        const float* __restrict__ w_off,
                        const float* __restrict__ b_off,
                        float* __restrict__ offs,
                        float* __restrict__ maskp,
                        int H, int W) {
    int pix = blockIdx.x;
    int b = pix / (H * W);
    int rem = pix - b * H * W;
    int h = rem / W, w = rem - (rem / W) * W;
    int c = threadIdx.x;
    int lane = c & 31, wv = c >> 5;

    float xv[9];
#pragma unroll
    for (int t = 0; t < 9; ++t) {
        int yy = h - 1 + t / 3, xx = w - 1 + (t % 3);
        bool ok = (yy >= 0) && (yy < H) && (xx >= 0) && (xx < W);
        xv[t] = ok ? x[((b * CH + c) * H + yy) * W + xx] : 0.0f;
    }
    __shared__ float wred[8];
    for (int oc = 0; oc < 27; ++oc) {
        const float* wp = w_off + (oc * CH + c) * 9;
        float p = 0.0f;
#pragma unroll
        for (int t = 0; t < 9; ++t) p += xv[t] * wp[t];
        for (int off = 16; off > 0; off >>= 1) p += __shfl_down(p, off, 32);
        if (lane == 0) wred[wv] = p;
        __syncthreads();
        if (c == 0) {
            float v = b_off[oc];
            for (int i = 0; i < 8; ++i) v += wred[i];
            if (oc < 18) offs[((b * 18 + oc) * H + h) * W + w] = v;
            else maskp[((b * 9 + (oc - 18)) * H + h) * W + w] = 1.0f / (1.0f + __expf(-v));
        }
        __syncthreads();
    }
}

// --------------------------------------------------------------------------
// Fused deformable-im2col + GEMM via WMMA (bf16 in / f32 acc).
//   feat[o][n] = sum_K wq[o][K] * cols[K][n],  K = tap*256 + c
//   n = b*Ho*Wo + h*Wo + w.
// Grid: (N/16). Block: 256 thr = 8 waves; wave wv owns 32 rows
//   (two 16-row M-tiles at wv*32, wv*32+16), so a block covers all M=256.
// Per sync round a 64(K) x 16(N) bf16 B-tile is staged in LDS and each wave
// issues 4 WMMAs (2 M-tiles x 2 K-subtiles).
// --------------------------------------------------------------------------
__global__ __launch_bounds__(256)
void mdcn_wmma_kernel(const float* __restrict__ xsrc,            // [4][256][Hs][Ws]
                      const unsigned short* __restrict__ wq,     // [256][2304] bf16
                      const float* __restrict__ offs,            // [4][18][Hoff][Woff]
                      const float* __restrict__ maskp,           // [4][9][Hoff][Woff]
                      float* __restrict__ feat,                  // [256][N]
                      int Hs, int Ws, int Ho, int Wo,
                      int Hoff, int Woff, int stride, int ostep) {
    const int N = 4 * Ho * Wo;
    const int tid = threadIdx.x;
    const int lane = tid & 31;
    const int wv = tid >> 5;
    const int khalf = lane >> 4;
    const int l16 = lane & 15;
    const int ntile = blockIdx.x;
    const int mbase = wv * 32;
    const int HsWs = Hs * Ws;

    __shared__ __align__(4) unsigned short Bs[16][66]; // [n][K 0..63], padded row
    __shared__ int   s_base[16][4];
    __shared__ float s_wt[16][4];

    v8f acc0 = {}, acc1 = {};

    for (int kt = 0; kt < KTOT / 64; ++kt) {            // 36 rounds of K=64
        const int tap = kt >> 2;
        const int c0 = (kt & 3) << 6;
        __syncthreads();
        if ((kt & 3) == 0) {
            if (tid < 16) {
                // new tap: recompute bilinear params for the 16 columns
                int n = ntile * 16 + tid;
                int b = n / (Ho * Wo);
                int rem = n - b * (Ho * Wo);
                int hh = rem / Wo, ww = rem - (rem / Wo) * Wo;
                int ho = hh * ostep, wo = ww * ostep;
                float offy = offs[((b * 18 + tap * 2) * Hoff + ho) * Woff + wo];
                float offx = offs[((b * 18 + tap * 2 + 1) * Hoff + ho) * Woff + wo];
                float m    = maskp[((b * 9 + tap) * Hoff + ho) * Woff + wo];
                float py = (float)(hh * stride - 1 + tap / 3) + offy;
                float px = (float)(ww * stride - 1 + tap % 3) + offx;
                float y0f = floorf(py), x0f = floorf(px);
                int bbase = b * CH * HsWs;
#pragma unroll
                for (int j = 0; j < 4; ++j) {
                    float yi = y0f + (float)(j >> 1);
                    float xi = x0f + (float)(j & 1);
                    float wgt = (1.0f - fabsf(py - yi)) * (1.0f - fabsf(px - xi));
                    bool valid = (yi >= 0.0f) && (yi <= (float)(Hs - 1)) &&
                                 (xi >= 0.0f) && (xi <= (float)(Ws - 1));
                    int yc = (int)fminf(fmaxf(yi, 0.0f), (float)(Hs - 1));
                    int xc = (int)fminf(fmaxf(xi, 0.0f), (float)(Ws - 1));
                    s_base[tid][j] = bbase + yc * Ws + xc;
                    s_wt[tid][j] = valid ? wgt * m : 0.0f;
                }
            }
            __syncthreads();
        }
        // stage 64(K) x 16(N) bf16 B-tile (1024 elems / 256 threads)
        for (int e = tid; e < 1024; e += 256) {
            int cl = e >> 4, nl = e & 15;
            int coff = (c0 + cl) * HsWs;
            float v = s_wt[nl][0] * xsrc[s_base[nl][0] + coff]
                    + s_wt[nl][1] * xsrc[s_base[nl][1] + coff]
                    + s_wt[nl][2] * xsrc[s_base[nl][2] + coff]
                    + s_wt[nl][3] * xsrc[s_base[nl][3] + coff];
            Bs[nl][cl] = f2bf(v);
        }
        __syncthreads();
        // B fragments for the two 32-K subtiles (ISA 7.12.2 16-bit layout)
        union U { v16bf v; unsigned int u[8]; };
        U Bf0, Bf1;
#pragma unroll
        for (int p = 0; p < 8; ++p) {
            int kl = (p < 4) ? (khalf * 8 + 2 * p) : (16 + khalf * 8 + 2 * (p - 4));
            Bf0.u[p] = *(const unsigned int*)(&Bs[l16][kl]);
            Bf1.u[p] = *(const unsigned int*)(&Bs[l16][32 + kl]);
        }
        const unsigned short* wbase = wq + kt * 64;
#pragma unroll
        for (int m = 0; m < 2; ++m) {
            const unsigned short* wrow = wbase + (mbase + m * 16 + l16) * KTOT;
            U A0, A1;
#pragma unroll
            for (int p = 0; p < 8; ++p) {
                int kl = (p < 4) ? (khalf * 8 + 2 * p) : (16 + khalf * 8 + 2 * (p - 4));
                A0.u[p] = *(const unsigned int*)(wrow + kl);
                A1.u[p] = *(const unsigned int*)(wrow + 32 + kl);
            }
            v8f& acc = m ? acc1 : acc0;
            acc = __builtin_amdgcn_wmma_f32_16x16x32_bf16(
                false, A0.v, false, Bf0.v, (short)0, acc, false, false);
            acc = __builtin_amdgcn_wmma_f32_16x16x32_bf16(
                false, A1.v, false, Bf1.v, (short)0, acc, false, false);
        }
    }
    // D layout: VGPR r -> M = r + 8*khalf, N = lane%16
    int n = ntile * 16 + l16;
#pragma unroll
    for (int r = 0; r < 8; ++r) {
        feat[(mbase + r + 8 * khalf) * N + n]      = acc0[r];
        feat[(mbase + 16 + r + 8 * khalf) * N + n] = acc1[r];
    }
}

// --------------------------------------------------------------------------
// Per-(b,c) sum / sumsq over HW of feat[256][4*HW]
// --------------------------------------------------------------------------
__global__ __launch_bounds__(256)
void chan_stats_kernel(const float* __restrict__ feat,
                       float* __restrict__ csum, float* __restrict__ csq, int HW) {
    int bc = blockIdx.x;          // b*256 + c
    int b = bc >> 8, c = bc & 255;
    const float* p = feat + c * (4 * HW) + b * HW;
    float s = 0.0f, q = 0.0f;
    for (int i = threadIdx.x; i < HW; i += 256) { float v = p[i]; s += v; q += v * v; }
    int lane = threadIdx.x & 31, wv = threadIdx.x >> 5;
    for (int off = 16; off > 0; off >>= 1) { s += __shfl_down(s, off, 32); q += __shfl_down(q, off, 32); }
    __shared__ float rs[8], rq[8];
    if (lane == 0) { rs[wv] = s; rq[wv] = q; }
    __syncthreads();
    if (threadIdx.x == 0) {
        float ts = 0.0f, tq = 0.0f;
        for (int i = 0; i < 8; ++i) { ts += rs[i]; tq += rq[i]; }
        csum[bc] = ts; csq[bc] = tq;
    }
}

// --------------------------------------------------------------------------
// GroupNorm stats (mu, rstd per (b,g)) + scale-attention scalar per b.
// One block (256 thr) per batch element.
// --------------------------------------------------------------------------
__global__ __launch_bounds__(256)
void gn_finalize_kernel(const float* __restrict__ csum, const float* __restrict__ csq,
                        const float* __restrict__ gw, const float* __restrict__ gb,
                        const float* __restrict__ sa_w, const float* __restrict__ sa_b,
                        float* __restrict__ mu, float* __restrict__ rstd,
                        float* __restrict__ attn, int HW) {
    int b = blockIdx.x;
    int c = threadIdx.x;
    __shared__ float ss[256], sq[256], smu[16], srstd[16], red[256];
    ss[c] = csum[b * 256 + c];
    sq[c] = csq[b * 256 + c];
    __syncthreads();
    if (c < 16) {
        float s = 0.0f, q = 0.0f;
        for (int j = 0; j < 16; ++j) { s += ss[c * 16 + j]; q += sq[c * 16 + j]; }
        float cnt = 16.0f * (float)HW;
        float m = s / cnt;
        float var = q / cnt - m * m;
        smu[c] = m;
        srstd[c] = rsqrtf(var + 1e-5f);
        mu[b * 16 + c] = m;
        rstd[b * 16 + c] = srstd[c];
    }
    __syncthreads();
    int g = c >> 4;
    float cm = ss[c] / (float)HW;
    float pc = (cm - smu[g]) * srstd[g] * gw[c] + gb[c];
    red[c] = pc * sa_w[c];
    __syncthreads();
    for (int sft = 128; sft > 0; sft >>= 1) {
        if (c < sft) red[c] += red[c + sft];
        __syncthreads();
    }
    if (c == 0) attn[b] = hsig(fmaxf(red[0] + sa_b[0], 0.0f));
}

// --------------------------------------------------------------------------
// s (layout [B][C][H][W]) accumulation of group-normed feature * attn.
// mode 0: store (mid); 1: add (low); 2: add with align-corner bilinear
// upsample from (Hf,Wf) -> (H,W). GN is per-channel affine, so interpolating
// pre-GN values equals GN-then-resize.
// --------------------------------------------------------------------------
__global__ __launch_bounds__(256)
void accumulate_kernel(float* __restrict__ s, const float* __restrict__ feat,
                       const float* __restrict__ mu, const float* __restrict__ rstd,
                       const float* __restrict__ gw, const float* __restrict__ gb,
                       const float* __restrict__ attn,
                       int H, int W, int Hf, int Wf, int mode) {
    int idx = blockIdx.x * 256 + threadIdx.x;
    int total = 4 * CH * H * W;
    if (idx >= total) return;
    int ww = idx % W; int t = idx / W;
    int hh = t % H; t /= H;
    int c = t % CH; int b = t / CH;
    int g = c >> 4;
    float m = mu[b * 16 + g], r = rstd[b * 16 + g];
    float ga = gw[c], be = gb[c];
    int Nf = 4 * Hf * Wf;
    float f;
    if (mode < 2) {
        f = feat[c * Nf + (b * Hf + hh) * Wf + ww];
    } else {
        float sy = (H > 1) ? (float)hh * (float)(Hf - 1) / (float)(H - 1) : 0.0f;
        float sx = (W > 1) ? (float)ww * (float)(Wf - 1) / (float)(W - 1) : 0.0f;
        int y0 = (int)floorf(sy); int y1 = (y0 + 1 < Hf) ? y0 + 1 : Hf - 1;
        int x0 = (int)floorf(sx); int x1 = (x0 + 1 < Wf) ? x0 + 1 : Wf - 1;
        float wy = sy - (float)y0, wx = sx - (float)x0;
        const float* fp = feat + c * Nf + b * Hf * Wf;
        float f00 = fp[y0 * Wf + x0], f01 = fp[y0 * Wf + x1];
        float f10 = fp[y1 * Wf + x0], f11 = fp[y1 * Wf + x1];
        f = (f00 * (1.0f - wx) + f01 * wx) * (1.0f - wy)
          + (f10 * (1.0f - wx) + f11 * wx) * wy;
    }
    float val = ((f - m) * r * ga + be) * attn[b];
    if (mode == 0) s[idx] = val; else s[idx] += val;
}

// --------------------------------------------------------------------------
// DyReLU global pool: p2[b*256+c] = mean(s/cnt) over HW
// --------------------------------------------------------------------------
__global__ __launch_bounds__(256)
void pool_s_kernel(const float* __restrict__ s, float* __restrict__ p2,
                   int HW, float inv_cnt) {
    int bc = blockIdx.x;
    const float* p = s + (size_t)bc * HW;
    float acc = 0.0f;
    for (int i = threadIdx.x; i < HW; i += 256) acc += p[i];
    int lane = threadIdx.x & 31, wv = threadIdx.x >> 5;
    for (int off = 16; off > 0; off >>= 1) acc += __shfl_down(acc, off, 32);
    __shared__ float red[8];
    if (lane == 0) red[wv] = acc;
    __syncthreads();
    if (threadIdx.x == 0) {
        float t = 0.0f;
        for (int i = 0; i < 8; ++i) t += red[i];
        p2[bc] = t * inv_cnt / (float)HW;
    }
}

// --------------------------------------------------------------------------
// DyReLU coefficients: h = relu(p@w1^T+b1) (64); co = hsig(h@w2^T+b2)-0.5
// split into a1*2+1, c1, a2*2, c2 (each [B][256]). One block per b.
// --------------------------------------------------------------------------
__global__ __launch_bounds__(256)
void dyrelu_coef_kernel(const float* __restrict__ p2,
                        const float* __restrict__ w1, const float* __restrict__ b1,
                        const float* __restrict__ w2, const float* __restrict__ b2,
                        float* __restrict__ a1, float* __restrict__ c1,
                        float* __restrict__ a2, float* __restrict__ c2) {
    int b = blockIdx.x, t = threadIdx.x;
    __shared__ float sp[256], sh[64];
    sp[t] = p2[b * 256 + t];
    __syncthreads();
    if (t < 64) {
        float acc = b1[t];
        for (int c = 0; c < 256; ++c) acc += sp[c] * w1[t * 256 + c];
        sh[t] = fmaxf(acc, 0.0f);
    }
    __syncthreads();
#pragma unroll
    for (int i = 0; i < 4; ++i) {
        int j = i * 256 + t;
        float acc = b2[j];
        for (int k = 0; k < 64; ++k) acc += sh[k] * w2[j * 64 + k];
        float v = hsig(acc) - 0.5f;
        if (i == 0) a1[b * 256 + t] = v * 2.0f + 1.0f;
        else if (i == 1) c1[b * 256 + t] = v;
        else if (i == 2) a2[b * 256 + t] = v * 2.0f;
        else c2[b * 256 + t] = v;
    }
}

// --------------------------------------------------------------------------
// Final: out = max(v*a1+c1, v*a2+c2), v = s/cnt.  Layout [B][C][H][W].
// --------------------------------------------------------------------------
__global__ __launch_bounds__(256)
void final_out_kernel(const float* __restrict__ s,
                      const float* __restrict__ a1, const float* __restrict__ c1,
                      const float* __restrict__ a2, const float* __restrict__ c2,
                      float* __restrict__ out, int HW, float inv_cnt) {
    int idx = blockIdx.x * 256 + threadIdx.x;
    int total = 4 * CH * HW;
    if (idx >= total) return;
    int bc = idx / HW;
    float v = s[idx] * inv_cnt;
    out[idx] = fmaxf(v * a1[bc] + c1[bc], v * a2[bc] + c2[bc]);
}

// ===========================================================================
extern "C" void kernel_launch(void* const* d_in, const int* in_sizes, int n_in,
                              void* d_out, int out_size, void* d_ws, size_t ws_size,
                              hipStream_t stream) {
    (void)in_sizes; (void)n_in; (void)out_size; (void)ws_size;
    const float* x[3]   = {(const float*)d_in[0], (const float*)d_in[1], (const float*)d_in[2]};
    const float* w_off  = (const float*)d_in[3];
    const float* b_off  = (const float*)d_in[4];
    const float* w_conv[3] = {(const float*)d_in[5],   // high
                              (const float*)d_in[6],   // mid
                              (const float*)d_in[7]};  // low
    const float* gnw[3] = {(const float*)d_in[8],  (const float*)d_in[10], (const float*)d_in[12]};
    const float* gnb[3] = {(const float*)d_in[9],  (const float*)d_in[11], (const float*)d_in[13]};
    const float* sa_w   = (const float*)d_in[14];
    const float* sa_b   = (const float*)d_in[15];
    const float* dy_w1  = (const float*)d_in[16];
    const float* dy_b1  = (const float*)d_in[17];
    const float* dy_w2  = (const float*)d_in[18];
    const float* dy_b2  = (const float*)d_in[19];
    float* outp = (float*)d_out;

    // ---- carve workspace ----
    uintptr_t wp = (uintptr_t)d_ws;
    auto carve = [&](size_t bytes) -> void* {
        void* p = (void*)wp;
        wp += (bytes + 255) & ~(size_t)255;
        return p;
    };
    unsigned short* wq[3];                       // bf16 repacked weights (high, mid, low)
    for (int i = 0; i < 3; ++i) wq[i] = (unsigned short*)carve((size_t)CH * KTOT * 2);
    float* offs  = (float*)carve((size_t)4 * 18 * 64 * 64 * 4);
    float* maskp = (float*)carve((size_t)4 * 9 * 64 * 64 * 4);
    float* feat  = (float*)carve((size_t)CH * 4 * 64 * 64 * 4);
    float* sbuf  = (float*)carve((size_t)4 * CH * 64 * 64 * 4);
    float* csum  = (float*)carve(1024 * 4);
    float* csq   = (float*)carve(1024 * 4);
    float* mu    = (float*)carve(64 * 4);
    float* rstd  = (float*)carve(64 * 4);
    float* attn  = (float*)carve(16);
    float* p2    = (float*)carve(1024 * 4);
    float* ca1   = (float*)carve(1024 * 4);
    float* cc1   = (float*)carve(1024 * 4);
    float* ca2   = (float*)carve(1024 * 4);
    float* cc2   = (float*)carve(1024 * 4);

    // ---- repack conv weights to bf16 tap-major ----
    for (int i = 0; i < 3; ++i)
        prep_w_kernel<<<(CH * KTOT + 255) / 256, 256, 0, stream>>>(w_conv[i], wq[i]);

    const int dims[3] = {64, 32, 16};
    size_t out_off = 0;

    for (int lvl = 0; lvl < 3; ++lvl) {
        const int H = dims[lvl], W = dims[lvl], HW = H * W;
        const float cnt = 1.0f + (lvl > 0 ? 1.0f : 0.0f) + (lvl < 2 ? 1.0f : 0.0f);
        const int nElem = 4 * CH * HW;

        // 1) offsets + masks
        offset_conv_kernel<<<4 * HW, 256, 0, stream>>>(x[lvl], w_off, b_off, offs, maskp, H, W);

        // 2) mid branch (idx 1 in high/mid/low ordering)
        mdcn_wmma_kernel<<<4 * HW / 16, 256, 0, stream>>>(
            x[lvl], wq[1], offs, maskp, feat, H, W, H, W, H, W, /*stride*/1, /*ostep*/1);
        chan_stats_kernel<<<1024, 256, 0, stream>>>(feat, csum, csq, HW);
        gn_finalize_kernel<<<4, 256, 0, stream>>>(csum, csq, gnw[1], gnb[1], sa_w, sa_b,
                                                  mu, rstd, attn, HW);
        accumulate_kernel<<<(nElem + 255) / 256, 256, 0, stream>>>(
            sbuf, feat, mu, rstd, gnw[1], gnb[1], attn, H, W, H, W, /*mode*/0);

        // 3) low branch: src = x[lvl-1] (2H x 2W), stride 2
        if (lvl > 0) {
            int Hs = dims[lvl - 1];
            mdcn_wmma_kernel<<<4 * HW / 16, 256, 0, stream>>>(
                x[lvl - 1], wq[2], offs, maskp, feat, Hs, Hs, H, W, H, W, /*stride*/2, /*ostep*/1);
            chan_stats_kernel<<<1024, 256, 0, stream>>>(feat, csum, csq, HW);
            gn_finalize_kernel<<<4, 256, 0, stream>>>(csum, csq, gnw[2], gnb[2], sa_w, sa_b,
                                                      mu, rstd, attn, HW);
            accumulate_kernel<<<(nElem + 255) / 256, 256, 0, stream>>>(
                sbuf, feat, mu, rstd, gnw[2], gnb[2], attn, H, W, H, W, /*mode*/1);
        }

        // 4) high branch: src = x[lvl+1] (H/2 x W/2), offsets subsampled ::2,
        //    result upsampled back to (H,W)
        if (lvl < 2) {
            int Hh = dims[lvl + 1], HWh = Hh * Hh;
            mdcn_wmma_kernel<<<4 * HWh / 16, 256, 0, stream>>>(
                x[lvl + 1], wq[0], offs, maskp, feat, Hh, Hh, Hh, Hh, H, W, /*stride*/1, /*ostep*/2);
            chan_stats_kernel<<<1024, 256, 0, stream>>>(feat, csum, csq, HWh);
            gn_finalize_kernel<<<4, 256, 0, stream>>>(csum, csq, gnw[0], gnb[0], sa_w, sa_b,
                                                      mu, rstd, attn, HWh);
            accumulate_kernel<<<(nElem + 255) / 256, 256, 0, stream>>>(
                sbuf, feat, mu, rstd, gnw[0], gnb[0], attn, H, W, Hh, Hh, /*mode*/2);
        }

        // 5) DyReLU
        pool_s_kernel<<<1024, 256, 0, stream>>>(sbuf, p2, HW, 1.0f / cnt);
        dyrelu_coef_kernel<<<4, 256, 0, stream>>>(p2, dy_w1, dy_b1, dy_w2, dy_b2,
                                                  ca1, cc1, ca2, cc2);
        final_out_kernel<<<(nElem + 255) / 256, 256, 0, stream>>>(
            sbuf, ca1, cc1, ca2, cc2, outp + out_off, HW, 1.0f / cnt);
        out_off += (size_t)nElem;
    }
}
